// GraphBlock_20744692040123
// MI455X (gfx1250) — matrix operand order repaired and checked by reading, not compiled, not run on hardware
//
#include <hip/hip_runtime.h>
#include <hip/hip_bf16.h>
#include <math.h>

// ---------------------------------------------------------------------------
// Problem constants (from the reference)
// ---------------------------------------------------------------------------
#define BATCH    32
#define NNODE    128          // N = C_OUT
#define NODE_DIM 40
#define CL       4096         // C*L = 64*64
#define KCONV    384          // (GDEP+1)*N
#define MTILE    16           // columns per workgroup
#define LN_EPS   1e-5f

typedef __attribute__((ext_vector_type(16))) __bf16 v16bf;
typedef __attribute__((ext_vector_type(8)))  __bf16 v8bf;
typedef __attribute__((ext_vector_type(8)))  float  v8f;

// ---------------------------------------------------------------------------
// Kernel 1: normalized adjacency in bf16
//   a = (softmax(relu(nv1@nv2), axis=1) + I)/2   (softmax row-sum is exactly 1)
// grid = 128 rows, block = 128 threads (one per column)
// ---------------------------------------------------------------------------
__global__ __launch_bounds__(128)
void adj_kernel(const float* __restrict__ nv1, const float* __restrict__ nv2,
                __bf16* __restrict__ abf) {
    const int v = blockIdx.x;
    const int w = threadIdx.x;
    float s = 0.f;
#pragma unroll
    for (int d = 0; d < NODE_DIM; ++d)
        s += nv1[v * NODE_DIM + d] * nv2[d * NNODE + w];
    s = fmaxf(s, 0.f);

    __shared__ float buf[NNODE];
    buf[w] = s;
    __syncthreads();
    for (int off = 64; off > 0; off >>= 1) {
        if (w < off) buf[w] = fmaxf(buf[w], buf[w + off]);
        __syncthreads();
    }
    const float mx = buf[0];
    __syncthreads();

    const float e = __expf(s - mx);
    buf[w] = e;
    __syncthreads();
    for (int off = 64; off > 0; off >>= 1) {
        if (w < off) buf[w] = buf[w] + buf[w + off];
        __syncthreads();
    }
    const float sum = buf[0];

    abf[v * NNODE + w] = (__bf16)(0.5f * (e / sum + (v == w ? 1.f : 0.f)));
}

// ---------------------------------------------------------------------------
// Kernel 2: conv_w (128x384 f32) -> bf16, done once instead of per-workgroup
// ---------------------------------------------------------------------------
__global__ __launch_bounds__(256)
void convw_bf16_kernel(const float* __restrict__ w, __bf16* __restrict__ wbf) {
    const int i = blockIdx.x * 256 + threadIdx.x;   // 128*384 = 49152 elements
    wbf[i] = (__bf16)w[i];
}

// ---------------------------------------------------------------------------
// WMMA fragment helpers (wave32, v_wmma_f32_16x16x32_bf16)
// A 16x32 bf16: lane r=lane&15 holds row r; element e -> K = (e<8?e:e+8)+8*h
//   => per lane two contiguous 8-bf16 (16 B) runs -> two global_load_b128,
//      landing directly in the v16bf halves (no conversion VALU).
// B 32x16 bf16: lane n=lane&15 is col; element e -> K = (e<8?e:e+8)+8*h (same as A)
// C/D f32 16x16: element r -> row = r + 8*h, col = lane&15
// ---------------------------------------------------------------------------
__device__ inline v16bf load_a_frag_bf(const __bf16* __restrict__ A, int ld,
                                       int v0, int k0) {
    const int lane = threadIdx.x & 31;
    const int r = lane & 15, hl = lane >> 4;
    const __bf16* base = A + (size_t)(v0 + r) * ld + k0 + 8 * hl;
    const v8bf lo = *(const v8bf*)(base);        // K = k0+8h    .. +7
    const v8bf hi = *(const v8bf*)(base + 16);   // K = k0+16+8h .. +7
    v16bf o;
#pragma unroll
    for (int e = 0; e < 8; ++e) { o[e] = lo[e]; o[e + 8] = hi[e]; }
    return o;
}

#define BSTRIDE 136   // padded bf16 LDS stride (m*272B -> +4 banks per col)

__device__ inline v16bf load_b_frag_lds(const __bf16* __restrict__ Bl, int k0) {
    const int lane = threadIdx.x & 31;
    const int n = lane & 15, hl = lane >> 4;
    const __bf16* p = Bl + n * BSTRIDE + k0 + 8 * hl;
    v16bf o;
#pragma unroll
    for (int e = 0; e < 16; ++e) o[e] = p[e < 8 ? e : e + 8];
    return o;
}

__device__ inline v8f wmma_bf16(v16bf a, v16bf b, v8f c) {
    return __builtin_amdgcn_wmma_f32_16x16x32_bf16(
        /*neg_a=*/false, a, /*neg_b=*/false, b,
        /*c_mod=*/(short)0, c, /*reuse_a=*/false, /*reuse_b=*/false);
}

// ---------------------------------------------------------------------------
// Kernel 3: fully fused mixprop + 1x1 conv + GELU + residual + transpose + LN
// grid = BATCH * (CL/MTILE) = 8192 workgroups; block = 256 threads (8 waves)
// Wave i owns output-node band [16i, 16i+16) for this 16-column tile.
// ---------------------------------------------------------------------------
#define FSTRIDE 132   // padded f32 LDS stride

__global__ __launch_bounds__(256)
void fused_graphblock(const float*  __restrict__ x,      // (B, 128, CL)
                      const __bf16* __restrict__ abf,    // (128, 128) norm adj
                      const __bf16* __restrict__ wbf,    // (128, 384) conv w
                      const float*  __restrict__ conv_b, // (128,)
                      const float*  __restrict__ ln_w,   // (128,)
                      const float*  __restrict__ ln_b,   // (128,)
                      float* __restrict__ out) {         // (B, CL, 128)
    const int tid = threadIdx.x;
    const int b   = blockIdx.x / (CL / MTILE);
    const int m0  = (blockIdx.x % (CL / MTILE)) * MTILE;

    __shared__ __bf16 xb [MTILE][BSTRIDE];   // x tile  (bf16, [col][node])
    __shared__ __bf16 h1b[MTILE][BSTRIDE];   // h1 tile
    __shared__ __bf16 h2b[MTILE][BSTRIDE];   // h2 tile
    __shared__ float  xf [MTILE][FSTRIDE];   // x tile  (f32, for mix/residual)
    __shared__ float  yt [MTILE][FSTRIDE];   // pre-LN output tile
    __shared__ float  red [MTILE][16];
    __shared__ float  red2[MTILE][16];
    __shared__ float  mean_s[MTILE], rstd_s[MTILE];

    const int wv   = tid >> 5;            // wave id 0..7
    const int v0   = wv * 16;             // row band
    const int lane = tid & 31;
    const int mcol = lane & 15;
    const int hl   = lane >> 4;

    // Warm the wave's conv-weight rows into cache while x streams in
    // (lowers to global_prefetch_b8).
    __builtin_prefetch(wbf + (size_t)(v0 + mcol) * KCONV + hl * 192, 0, 1);

    // ---- load x tile: 128 nodes x 16 cols, transposed into [col][node] ----
    const float* xg = x + (size_t)b * NNODE * CL + m0;
#pragma unroll
    for (int i = tid; i < NNODE * 4; i += 256) {      // 512 float4 loads
        const int w = i >> 2;                         // node
        const int q = i & 3;                          // col quad
        const float4 v4 = *(const float4*)(xg + (size_t)w * CL + q * 4);
        xf[q * 4 + 0][w] = v4.x; xb[q * 4 + 0][w] = (__bf16)v4.x;
        xf[q * 4 + 1][w] = v4.y; xb[q * 4 + 1][w] = (__bf16)v4.y;
        xf[q * 4 + 2][w] = v4.z; xb[q * 4 + 2][w] = (__bf16)v4.z;
        xf[q * 4 + 3][w] = v4.w; xb[q * 4 + 3][w] = (__bf16)v4.w;
    }
    __syncthreads();

    // adjacency A-fragments stay resident; reused by both propagation GEMMs
    v16bf afrag[4];
#pragma unroll
    for (int kb = 0; kb < 4; ++kb)
        afrag[kb] = load_a_frag_bf(abf, NNODE, v0, kb * 32);

    // ---- h1 = 0.5*x + 0.5*(a @ x) ----
    v8f acc = {};
#pragma unroll
    for (int kb = 0; kb < 4; ++kb)
        acc = wmma_bf16(afrag[kb], load_b_frag_lds(&xb[0][0], kb * 32), acc);
#pragma unroll
    for (int r = 0; r < 8; ++r) {
        const int row = v0 + r + 8 * hl;
        const float hv = 0.5f * xf[mcol][row] + 0.5f * acc[r];
        h1b[mcol][row] = (__bf16)hv;
    }
    __syncthreads();

    // ---- h2 = 0.5*x + 0.5*(a @ h1) ----
    v8f acc2 = {};
#pragma unroll
    for (int kb = 0; kb < 4; ++kb)
        acc2 = wmma_bf16(afrag[kb], load_b_frag_lds(&h1b[0][0], kb * 32), acc2);
#pragma unroll
    for (int r = 0; r < 8; ++r) {
        const int row = v0 + r + 8 * hl;
        const float hv = 0.5f * xf[mcol][row] + 0.5f * acc2[r];
        h2b[mcol][row] = (__bf16)hv;
    }
    __syncthreads();

    // ---- conv: out = W @ [x; h1; h2]  (K = 384), bf16 A-frags from global ----
    v8f accc = {};
#pragma unroll
    for (int kb = 0; kb < 12; ++kb) {
        const __bf16* src = (kb < 4) ? &xb[0][0] : (kb < 8) ? &h1b[0][0] : &h2b[0][0];
        const int klocal = (kb & 3) * 32;
        const v16bf wfrag = load_a_frag_bf(wbf, KCONV, v0, kb * 32);
        accc = wmma_bf16(wfrag, load_b_frag_lds(src, klocal), accc);
    }

    // ---- bias + exact GELU + residual -> yt[col][node] ----
#pragma unroll
    for (int r = 0; r < 8; ++r) {
        const int o = v0 + r + 8 * hl;
        const float v = accc[r] + conv_b[o];
        const float g = 0.5f * v * (1.0f + erff(v * 0.70710678118654752f));
        yt[mcol][o] = xf[mcol][o] + g;
    }
    __syncthreads();

    // ---- LayerNorm over node dim (128) per column ----
    const int col = tid & 15;
    const int sub = tid >> 4;             // 16 partial groups of 8 rows
    {
        float s = 0.f, s2 = 0.f;
#pragma unroll
        for (int j = 0; j < 8; ++j) {
            const float v = yt[col][sub * 8 + j];
            s += v; s2 += v * v;
        }
        red[col][sub] = s; red2[col][sub] = s2;
    }
    __syncthreads();
    if (sub == 0) {
        float S = 0.f, S2 = 0.f;
#pragma unroll
        for (int j = 0; j < 16; ++j) { S += red[col][j]; S2 += red2[col][j]; }
        const float mean = S * (1.0f / NNODE);
        const float var  = S2 * (1.0f / NNODE) - mean * mean;
        mean_s[col] = mean;
        rstd_s[col] = rsqrtf(var + LN_EPS);
    }
    __syncthreads();

    // ---- store y[b, m, n], coalesced in n ----
    float* og = out + ((size_t)b * CL + m0) * NNODE;
#pragma unroll
    for (int i = tid; i < MTILE * NNODE; i += 256) {
        const int ml = i >> 7;
        const int n  = i & 127;
        const float v = (yt[ml][n] - mean_s[ml]) * rstd_s[ml] * ln_w[n] + ln_b[n];
        og[(size_t)ml * NNODE + n] = v;
    }
}

// ---------------------------------------------------------------------------
extern "C" void kernel_launch(void* const* d_in, const int* in_sizes, int n_in,
                              void* d_out, int out_size, void* d_ws, size_t ws_size,
                              hipStream_t stream) {
    const float* x      = (const float*)d_in[0];
    const float* nv1    = (const float*)d_in[1];
    const float* nv2    = (const float*)d_in[2];
    const float* conv_w = (const float*)d_in[3];
    const float* conv_b = (const float*)d_in[4];
    const float* ln_w   = (const float*)d_in[5];
    const float* ln_b   = (const float*)d_in[6];
    float* out = (float*)d_out;

    // workspace: bf16 adjacency (32 KB) then bf16 conv weights (96 KB)
    __bf16* abf = (__bf16*)d_ws;
    __bf16* wbf = abf + NNODE * NNODE;

    adj_kernel<<<dim3(NNODE), dim3(NNODE), 0, stream>>>(nv1, nv2, abf);
    convw_bf16_kernel<<<dim3(NNODE * KCONV / 256), dim3(256), 0, stream>>>(conv_w, wbf);

    const int nblocks = BATCH * (CL / MTILE);  // 8192
    fused_graphblock<<<dim3(nblocks), dim3(256), 0, stream>>>(
        x, abf, wbf, conv_b, ln_w, ln_b, out);
}
